// MS_CrossIIWA_77610059038870
// MI455X (gfx1250) — matrix operand order repaired and checked by reading, not compiled, not run on hardware
//
#include <hip/hip_runtime.h>
#include <hip/hip_bf16.h>

// ---------------------------------------------------------------------------
// MS_CrossIIWA for MI455X (gfx1250, wave32, WMMA).
// All heavy GEMMs use v_wmma_f32_16x16x32_f16 (f16 in, f32 accumulate).
// ---------------------------------------------------------------------------

typedef __attribute__((ext_vector_type(16))) _Float16 v16h;
typedef __attribute__((ext_vector_type(8)))  float    v8f;

#define NB      2      // batch
#define CD      256    // channels
#define NHEADS  8
#define HDIM    32
#define NWIN    1024   // query windows per batch (32x32)
#define NKWIN   1344   // key windows per batch (1024+256+64)
#define WTOK    64     // tokens per window (8x8)

__device__ __forceinline__ v8f wmma_f16(v16h a, v16h b, v8f c) {
    return __builtin_amdgcn_wmma_f32_16x16x32_f16(
        /*neg_a=*/false, a, /*neg_b=*/false, b,
        /*c_mod=*/(short)0, c, /*reuse_a=*/false, /*reuse_b=*/false);
}

// A-fragment (16x32 f16), row-major source [rows][rowstride], rows at Mbase..Mbase+15,
// cols at colbase..colbase+31. Per lane: K covers (lane/16)*16 + e (contiguous, matches B).
__device__ __forceinline__ v16h load_a_frag(const _Float16* src, int rowstride,
                                            int Mbase, int colbase, int lane) {
    int r  = lane & 15;
    int hi = lane >> 4;
    const _Float16* row = src + (Mbase + r) * rowstride + colbase;
    v16h a;
#pragma unroll
    for (int v = 0; v < 8; ++v) {
        int k = (hi << 4) + (v << 1);
        a[2 * v]     = row[k];
        a[2 * v + 1] = row[k + 1];
    }
    return a;
}

// B-fragment (32x16 f16) where B[k][n] = src[(nbase+n)*rowstride + colbase + k].
// Per lane: n = lane%16, K covers (lane/16)*16 + e  -> 16 contiguous halves.
__device__ __forceinline__ v16h load_b_frag(const _Float16* src, int rowstride,
                                            int nbase, int colbase, int lane) {
    int n  = lane & 15;
    int hi = lane >> 4;
    const _Float16* p = src + (nbase + n) * rowstride + colbase + (hi << 4);
    v16h b;
#pragma unroll
    for (int e = 0; e < 16; ++e) b[e] = p[e];
    return b;
}

// B-fragment for attn @ V: B[k=token][n=dd] = sVh[token*256 + h*32 + n] (strided gather)
__device__ __forceinline__ v16h load_bv_frag(const _Float16* sVh, int h, int ni,
                                             int kb, int lane) {
    int n  = ni * 16 + (lane & 15);
    int hi = lane >> 4;
    v16h b;
#pragma unroll
    for (int e = 0; e < 16; ++e) {
        int ktok = kb * 32 + (hi << 4) + e;
        b[e] = sVh[ktok * CD + h * HDIM + n];
    }
    return b;
}

// ---------------------------------------------------------------------------
// K0a: convert the four 256x256 weight matrices to f16 (row-major, original layout)
__global__ void k0_convert(const float* __restrict__ Wq, const float* __restrict__ Wk,
                           const float* __restrict__ Wv, const float* __restrict__ Wp,
                           _Float16* Wq16, _Float16* Wk16, _Float16* Wv16, _Float16* Wp16) {
    int i = blockIdx.x * 256 + threadIdx.x;
    if (i < CD * CD) {
        Wq16[i] = (_Float16)Wq[i];
        Wk16[i] = (_Float16)Wk[i];
        Wv16[i] = (_Float16)Wv[i];
        Wp16[i] = (_Float16)Wp[i];
    }
}

// K0b: expand relative position bias: biasb[h][n][m] (8 x 64 x 64 f32)
__global__ void k0_bias(const float* __restrict__ tbl, float* __restrict__ biasb) {
    int id = blockIdx.x * 256 + threadIdx.x;
    if (id < 64 * 64) {
        int n = id >> 6, m = id & 63;
        int r0 = (n >> 3) - (m >> 3) + 7;
        int r1 = (n & 7) - (m & 7) + 7;
        int rpi = r0 * 15 + r1;                  // (2*8-1) stride
#pragma unroll
        for (int h = 0; h < NHEADS; ++h)
            biasb[h * 4096 + n * 64 + m] = tbl[rpi * NHEADS + h];
    }
}

// ---------------------------------------------------------------------------
// K1: per-window query mean  qg[b, win, c]
__global__ void k1_qg(const float* __restrict__ q, float* __restrict__ qg) {
    int wid = blockIdx.x;                 // b*1024 + win
    int b = wid >> 10, win = wid & 1023;
    int wy = win >> 5, wx = win & 31;
    int t = threadIdx.x;                  // channel
    float s = 0.f;
    for (int r = 0; r < WTOK; ++r) {
        int iy = r >> 3, ix = r & 7;
        size_t g = ((size_t)((b * 256 + wy * 8 + iy) * 256 + (wx * 8 + ix))) * CD + t;
        s += q[g];
    }
    qg[(size_t)wid * CD + t] = s * (1.0f / 64.0f);
}

// ---------------------------------------------------------------------------
// K2: sim[b, n, m] = qg[b,n,:] . kg[b,m,:]   (only feeds the argmax; VALU f32)
__global__ void k2_sim(const float* __restrict__ qg,
                       const float* __restrict__ kg1, const float* __restrict__ kg2,
                       const float* __restrict__ kg3, float* __restrict__ sim) {
    __shared__ float qr[CD];
    int wid = blockIdx.x;                 // b*1024 + n
    int b = wid >> 10;
    int t = threadIdx.x;
    qr[t] = qg[(size_t)wid * CD + t];
    __syncthreads();
    for (int m = t; m < NKWIN; m += 256) {
        const float* kr;
        if (m < 1024)      kr = kg1 + (size_t)(b * 1024 + m) * CD;
        else if (m < 1280) kr = kg2 + (size_t)(b * 256 + (m - 1024)) * CD;
        else               kr = kg3 + (size_t)(b * 64 + (m - 1280)) * CD;
        float d = 0.f;
        for (int c = 0; c < CD; ++c) d += qr[c] * kr[c];
        sim[(size_t)wid * NKWIN + m] = d;
    }
}

// ---------------------------------------------------------------------------
// K3: greedy row-by-row masked argmax (sequential scan; first-index tie-break)
__global__ void k3_assign(const float* __restrict__ sim, int* __restrict__ idx) {
    __shared__ int used[NKWIN];
    __shared__ float rv[256];
    __shared__ int   ri[256];
    int b = blockIdx.x, t = threadIdx.x;
    for (int m = t; m < NKWIN; m += 256) used[m] = 0;
    __syncthreads();
    for (int n = 0; n < NWIN; ++n) {
        const float* row = sim + ((size_t)b * NWIN + n) * NKWIN;
        float bval = -3.4e38f; int bm = 0x7fffffff;
        for (int m = t; m < NKWIN; m += 256) {
            if (!used[m]) {
                float v = row[m];
                if (v > bval || (v == bval && m < bm)) { bval = v; bm = m; }
            }
        }
        rv[t] = bval; ri[t] = bm;
        __syncthreads();
        for (int s = 128; s > 0; s >>= 1) {
            if (t < s) {
                if (rv[t + s] > rv[t] || (rv[t + s] == rv[t] && ri[t + s] < ri[t])) {
                    rv[t] = rv[t + s]; ri[t] = ri[t + s];
                }
            }
            __syncthreads();
        }
        if (t == 0) { idx[b * NWIN + n] = ri[0]; used[ri[0]] = 1; }
        __syncthreads();
    }
}

// ---------------------------------------------------------------------------
// K4: fused per-window attention. 1 block = 1 query window; 8 waves = 8 heads.
// Dynamic LDS layout (bytes):
//   [0,32K)   sQ16  64x256 f16   (q window)     \  aliased by sS (8 heads x
//   [32K,64K) sK16  64x256 f16   (k window)     /  64x64 f16) after projections
//   [64K,96K)  sQh  64x256 f16 (scaled)
//   [96K,128K) sKh  64x256 f16
//   [128K,160K)sVh  64x256 f16
__global__ __launch_bounds__(256) void k4_attn(
    const float* __restrict__ q,
    const float* __restrict__ k1, const float* __restrict__ k2, const float* __restrict__ k3,
    const int* __restrict__ idxbuf,
    const _Float16* __restrict__ Wq16, const _Float16* __restrict__ Wk16,
    const _Float16* __restrict__ Wv16,
    const float* __restrict__ bq, const float* __restrict__ bk, const float* __restrict__ bv,
    const float* __restrict__ biasb, float* __restrict__ out) {
    extern __shared__ char smem[];
    _Float16* sQ16 = (_Float16*)(smem);
    _Float16* sK16 = (_Float16*)(smem + 32768);
    _Float16* sQh  = (_Float16*)(smem + 65536);
    _Float16* sKh  = (_Float16*)(smem + 98304);
    _Float16* sVh  = (_Float16*)(smem + 131072);
    _Float16* sS   = (_Float16*)(smem);          // alias over sQ16+sK16

    const int t = threadIdx.x;
    const int wave = t >> 5, lane = t & 31;
    const int wid = blockIdx.x;
    const int b = wid >> 10, win = wid & 1023;
    const int wy = win >> 5, wx = win & 31;

    // gather source for assigned key window
    int kidx = idxbuf[b * NWIN + win];
    const float* ksrc;
    if (kidx < 1024)      ksrc = k1 + (size_t)(b * 1024 + kidx) * WTOK * CD;
    else if (kidx < 1280) ksrc = k2 + (size_t)(b * 256 + (kidx - 1024)) * WTOK * CD;
    else                  ksrc = k3 + (size_t)(b * 64 + (kidx - 1280)) * WTOK * CD;

    // stage q window + k window into LDS as f16 (coalesced: thread = channel)
    for (int r = 0; r < WTOK; ++r) {
        int iy = r >> 3, ix = r & 7;
        size_t gq = ((size_t)((b * 256 + wy * 8 + iy) * 256 + (wx * 8 + ix))) * CD + t;
        sQ16[r * CD + t] = (_Float16)q[gq];
        sK16[r * CD + t] = (_Float16)ksrc[r * CD + t];
    }
    __syncthreads();

    const float scale = 0.17677669529663687f;    // 32^-0.5

    // --- QKV projections: 64x256 = (64x256) @ (256x256)^T, WMMA f16 ---
    for (int p = 0; p < 3; ++p) {
        const _Float16* W   = (p == 0) ? Wq16 : (p == 1) ? Wk16 : Wv16;
        const float* bias   = (p == 0) ? bq : (p == 1) ? bk : bv;
        const _Float16* src = (p == 0) ? sQ16 : sK16;
        _Float16* dst       = (p == 0) ? sQh : (p == 1) ? sKh : sVh;
        const float sc      = (p == 0) ? scale : 1.0f;
        for (int tile = wave; tile < 64; tile += 8) {   // 4 M-tiles x 16 N-tiles
            int mi = tile >> 4, ni = tile & 15;
            v8f acc = {};
#pragma unroll
            for (int kb = 0; kb < 8; ++kb) {
                v16h a  = load_a_frag(src, CD, mi * 16, kb * 32, lane);
                v16h bb = load_b_frag(W, CD, ni * 16, kb * 32, lane);
                acc = wmma_f16(a, bb, acc);
            }
            int N  = ni * 16 + (lane & 15);
            int Mo = mi * 16 + ((lane >> 4) << 3);
            float bn = bias[N];
#pragma unroll
            for (int j = 0; j < 8; ++j)
                dst[(Mo + j) * CD + N] = (_Float16)((acc[j] + bn) * sc);
        }
    }
    __syncthreads();   // projections done; sQ16/sK16 dead -> sS may be written

    // --- per-head attention: wave w owns head w ---
    const int h = wave;
    _Float16* Sh = sS + h * 4096;                 // 64x64 f16
    const float* biash = biasb + h * 4096;

    // S = qh @ kh^T + bias   (K = 32 = head dim: single WMMA per tile)
#pragma unroll
    for (int mi = 0; mi < 4; ++mi) {
        v16h a = load_a_frag(sQh, CD, mi * 16, h * HDIM, lane);
#pragma unroll
        for (int ni = 0; ni < 4; ++ni) {
            v16h bb = load_b_frag(sKh, CD, ni * 16, h * HDIM, lane);
            v8f acc = {};
            acc = wmma_f16(a, bb, acc);
            int N  = ni * 16 + (lane & 15);
            int Mo = mi * 16 + ((lane >> 4) << 3);
#pragma unroll
            for (int j = 0; j < 8; ++j)
                Sh[(Mo + j) * 64 + N] = (_Float16)(acc[j] + biash[(Mo + j) * 64 + N]);
        }
    }

    // softmax over rows of Sh (wave-local: 2 rows per lane; f32 math)
    for (int rr = 0; rr < 2; ++rr) {
        int m = lane * 2 + rr;
        _Float16* row = Sh + m * 64;
        float mx = -3.0e38f;
        for (int j = 0; j < 64; ++j) { float v = (float)row[j]; mx = v > mx ? v : mx; }
        float s = 0.f;
        for (int j = 0; j < 64; ++j) {
            float e = __expf((float)row[j] - mx);
            s += e;
            row[j] = (_Float16)e;
        }
        float inv = 1.0f / s;
        for (int j = 0; j < 64; ++j) row[j] = (_Float16)((float)row[j] * inv);
    }

    // O = attn @ vh ; fuse window-reverse + residual (x = shortcut + attn_out)
#pragma unroll
    for (int mi = 0; mi < 4; ++mi) {
#pragma unroll
        for (int ni = 0; ni < 2; ++ni) {
            v8f acc = {};
#pragma unroll
            for (int kb = 0; kb < 2; ++kb) {
                v16h a  = load_a_frag(Sh, 64, mi * 16, kb * 32, lane);
                v16h bb = load_bv_frag(sVh, h, ni, kb, lane);
                acc = wmma_f16(a, bb, acc);
            }
            int N  = ni * 16 + (lane & 15);        // 0..31 within head
            int Mo = mi * 16 + ((lane >> 4) << 3); // token row
            int ch = h * HDIM + N;
#pragma unroll
            for (int j = 0; j < 8; ++j) {
                int m = Mo + j;
                int iy = m >> 3, ix = m & 7;
                size_t g = ((size_t)((b * 256 + wy * 8 + iy) * 256 + (wx * 8 + ix))) * CD + ch;
                out[g] = q[g] + acc[j];
            }
        }
    }
}

// ---------------------------------------------------------------------------
// K5: in-place  out = x + LayerNorm(x) @ Wp^T + bp   (block owns 64 token rows)
__global__ __launch_bounds__(256) void k5_final(float* __restrict__ xio,
                                                const float* __restrict__ gam,
                                                const float* __restrict__ bet,
                                                const _Float16* __restrict__ Wp16,
                                                const float* __restrict__ bp) {
    __shared__ _Float16 ln16[WTOK * CD];
    __shared__ float ps[WTOK][4], ps2[WTOK][4];
    __shared__ float mu[WTOK], rs[WTOK];
    int t = threadIdx.x;
    size_t base = (size_t)blockIdx.x * WTOK;     // first token row

    // LN statistics: 4 threads per row
    {
        int r = t >> 2, part = t & 3;
        const float* xr = xio + (base + r) * CD + part * 64;
        float s = 0.f, s2 = 0.f;
        for (int c = 0; c < 64; ++c) { float v = xr[c]; s += v; s2 += v * v; }
        ps[r][part] = s; ps2[r][part] = s2;
    }
    __syncthreads();
    if (t < WTOK) {
        float S  = ps[t][0] + ps[t][1] + ps[t][2] + ps[t][3];
        float S2 = ps2[t][0] + ps2[t][1] + ps2[t][2] + ps2[t][3];
        float m  = S * (1.0f / 256.0f);
        float var = S2 * (1.0f / 256.0f) - m * m;
        mu[t] = m;
        rs[t] = rsqrtf(var + 1e-5f);
    }
    __syncthreads();
    {
        float gc = gam[t], bc = bet[t];
        for (int r = 0; r < WTOK; ++r) {
            float v = xio[(base + r) * CD + t];
            ln16[r * CD + t] = (_Float16)((v - mu[r]) * rs[r] * gc + bc);
        }
    }
    __syncthreads();

    // GEMM: (64x256) @ Wp^T, WMMA f16; add bp + residual, store in place
    int wave = t >> 5, lane = t & 31;
    for (int tile = wave; tile < 64; tile += 8) {
        int mi = tile >> 4, ni = tile & 15;
        v8f acc = {};
#pragma unroll
        for (int kb = 0; kb < 8; ++kb) {
            v16h a  = load_a_frag(ln16, CD, mi * 16, kb * 32, lane);
            v16h bb = load_b_frag(Wp16, CD, ni * 16, kb * 32, lane);
            acc = wmma_f16(a, bb, acc);
        }
        int N  = ni * 16 + (lane & 15);
        int Mo = mi * 16 + ((lane >> 4) << 3);
        float bn = bp[N];
#pragma unroll
        for (int j = 0; j < 8; ++j) {
            size_t g = (base + Mo + j) * CD + N;
            xio[g] = xio[g] + acc[j] + bn;
        }
    }
}

// ---------------------------------------------------------------------------
extern "C" void kernel_launch(void* const* d_in, const int* in_sizes, int n_in,
                              void* d_out, int out_size, void* d_ws, size_t ws_size,
                              hipStream_t stream) {
    (void)in_sizes; (void)n_in; (void)out_size; (void)ws_size;
    const float* q   = (const float*)d_in[0];
    const float* k1  = (const float*)d_in[1];
    const float* k2  = (const float*)d_in[2];
    const float* k3  = (const float*)d_in[3];
    const float* kg1 = (const float*)d_in[4];
    const float* kg2 = (const float*)d_in[5];
    const float* kg3 = (const float*)d_in[6];
    const float* tbl = (const float*)d_in[7];
    const float* Wq  = (const float*)d_in[8];
    const float* bq  = (const float*)d_in[9];
    const float* Wk  = (const float*)d_in[10];
    const float* bk  = (const float*)d_in[11];
    const float* Wv  = (const float*)d_in[12];
    const float* bv  = (const float*)d_in[13];
    const float* lng = (const float*)d_in[14];
    const float* lnb = (const float*)d_in[15];
    const float* Wp  = (const float*)d_in[16];
    const float* bp  = (const float*)d_in[17];
    float* out = (float*)d_out;

    // workspace partition (~13.2 MB)
    char* w = (char*)d_ws;
    float* qg = (float*)w;                w += (size_t)NB * NWIN * CD * 4;      // 2 MB
    float* sim = (float*)w;               w += (size_t)NB * NWIN * NKWIN * 4;   // 11 MB
    int* idx = (int*)w;                   w += (size_t)NB * NWIN * 4;
    _Float16* Wq16 = (_Float16*)w;        w += (size_t)CD * CD * 2;
    _Float16* Wk16 = (_Float16*)w;        w += (size_t)CD * CD * 2;
    _Float16* Wv16 = (_Float16*)w;        w += (size_t)CD * CD * 2;
    _Float16* Wp16 = (_Float16*)w;        w += (size_t)CD * CD * 2;
    float* biasb = (float*)w;             w += (size_t)NHEADS * 64 * 64 * 4;

    k0_convert<<<256, 256, 0, stream>>>(Wq, Wk, Wv, Wp, Wq16, Wk16, Wv16, Wp16);
    k0_bias<<<16, 256, 0, stream>>>(tbl, biasb);
    k1_qg<<<NB * NWIN, 256, 0, stream>>>(q, qg);
    k2_sim<<<NB * NWIN, 256, 0, stream>>>(qg, kg1, kg2, kg3, sim);
    k3_assign<<<NB, 256, 0, stream>>>(sim, idx);
    k4_attn<<<NB * NWIN, 256, 163840, stream>>>(q, k1, k2, k3, idx,
                                                Wq16, Wk16, Wv16, bq, bk, bv,
                                                biasb, out);
    k5_final<<<(NB * 256 * 256) / WTOK, 256, 0, stream>>>(out, lng, lnb, Wp16, bp);
}
